// UNet3D_45964740001805
// MI455X (gfx1250) — matrix operand order repaired and checked by reading, not compiled, not run on hardware
//
#include <hip/hip_runtime.h>
#include <hip/hip_bf16.h>

// ---------------------------------------------------------------------------
// CDNA5 / gfx1250 sparse 3D UNet forward.
// All convs are implicit GEMMs on v_wmma_f32_16x16x32_bf16.
// - Per-stage working sets fit in MI455X's 192MB L2 -> A-fragments use direct
//   per-lane global loads (L2 hits) instead of LDS staging.
// - Weights pre-packed to bf16 in exact WMMA B lane order (contiguous 32B/lane).
// - Register blocking: each wave computes a 32-voxel x 32-cout tile, so each
//   (tap, kchunk) issues 2 A loads + 2 B loads -> 4 WMMAs.
// - All grid dims are powers of two: coordinates via shifts/masks (no idiv).
// ---------------------------------------------------------------------------

typedef __attribute__((ext_vector_type(16))) __bf16 v16bf;
typedef __attribute__((ext_vector_type(8)))  __bf16 v8bf;
typedef __attribute__((ext_vector_type(8)))  float  v8f;

#define GDIM 64
#define NPTS 131072
#define CIN0 96
#define ZDIM 480   // 96 + 128 + 256
#define NCLS 20
#define NCLSPAD 32
#define BN_EPS 1e-5f

// ------------------------------- utility kernels ---------------------------

__global__ void zero_u32_k(unsigned* __restrict__ p, size_t n) {
  size_t i = (size_t)blockIdx.x * blockDim.x + threadIdx.x;
  if (i < n) p[i] = 0u;
}

__global__ void scatter_k(const float* __restrict__ feats,
                          const int* __restrict__ coords,
                          __bf16* __restrict__ grid, float* __restrict__ mask,
                          int n, int c) {
  size_t idx = (size_t)blockIdx.x * blockDim.x + threadIdx.x;
  if (idx >= (size_t)n * c) return;
  int pi = (int)(idx / c), ci = (int)(idx % c);
  int vx = coords[pi * 3 + 0], vy = coords[pi * 3 + 1], vz = coords[pi * 3 + 2];
  int v = (vx * GDIM + vy) * GDIM + vz;
  grid[(size_t)v * c + ci] = (__bf16)feats[idx];
  if (ci == 0) mask[v] = 1.0f;
}

__global__ void pool_mask_k(const float* __restrict__ mi, float* __restrict__ mo,
                            int lgo) {
  int go = 1 << lgo, gi = go * 2;
  int idx = blockIdx.x * blockDim.x + threadIdx.x;
  if (idx >= go * go * go) return;
  int z = idx & (go - 1), y = (idx >> lgo) & (go - 1), x = idx >> (2 * lgo);
  float m = 0.0f;
#pragma unroll
  for (int i = 0; i < 2; ++i)
#pragma unroll
    for (int j = 0; j < 2; ++j)
#pragma unroll
      for (int k = 0; k < 2; ++k)
        m = fmaxf(m, mi[((2 * x + i) * gi + (2 * y + j)) * gi + (2 * z + k)]);
  mo[idx] = m;
}

__global__ void reduce_sum_k(const float* __restrict__ x, int n, float* __restrict__ out) {
  float s = 0.0f;
  for (size_t i = (size_t)blockIdx.x * blockDim.x + threadIdx.x; i < (size_t)n;
       i += (size_t)gridDim.x * blockDim.x)
    s += x[i];
  atomicAdd(out, s);
}

// Pack fp32 weights (tap, Cin, Cout) -> bf16 fragments in WMMA B lane order.
// Fragment (tap, kc, ct): lane L (n = L&15, khalf = L>>4) owns elements
// e=0..15 with k = kc*32 + khalf*8 + e (e<8), kc*32 + 16 + khalf*8 + (e-8) (e>=8).
__global__ void pack_w_k(const float* __restrict__ w, __bf16* __restrict__ pk,
                         int ntap, int cin, int cout, int coutpad) {
  int nkc = cin >> 5, nct = coutpad >> 4;
  size_t total = (size_t)ntap * nkc * nct * 512;
  size_t idx = (size_t)blockIdx.x * blockDim.x + threadIdx.x;
  if (idx >= total) return;
  int e = (int)(idx & 15);
  int L = (int)((idx >> 4) & 31);
  size_t frag = idx >> 9;
  int ct = (int)(frag % nct);
  int kc = (int)((frag / nct) % nkc);
  int tap = (int)(frag / ((size_t)nct * nkc));
  int n = L & 15, khalf = L >> 4;
  int k = kc * 32 + (e < 8 ? khalf * 8 + e : 16 + khalf * 8 + (e - 8));
  int co = ct * 16 + n;
  float v = (co < cout) ? w[((size_t)tap * cin + k) * cout + co] : 0.0f;
  pk[idx] = (__bf16)v;
}

// ------------------------------- conv / GEMM -------------------------------
// Each wave: 32 voxels x 32 cout, acc[2][2] of v8f. KSIZE==3 -> s1/p1,
// KSIZE==2 -> s2/p0, KSIZE==1 -> 1x1x1 conv or plain GEMM (classifier).
template <int KSIZE>
__global__ __launch_bounds__(64) void conv_wmma_k(
    const __bf16* __restrict__ in, const __bf16* __restrict__ wpk,
    float* __restrict__ out, const float* __restrict__ mask,
    int lgin, int lgout, int cin, int coutpad) {
  constexpr int STRIDE = (KSIZE == 2) ? 2 : 1;
  constexpr int PAD = (KSIZE == 3) ? 1 : 0;
  const int lane = threadIdx.x & 31;
  const int wave = threadIdx.x >> 5;
  const int nct = coutpad >> 4;
  const int ct0 = (blockIdx.y * 2 + wave) * 2;   // this wave's first ctile
  if (ct0 >= nct) return;                         // wave-uniform
  const int nkc = cin >> 5;
  const int m = lane & 15;
  const int khalf = lane >> 4;
  const int tile_base = blockIdx.x * 32;
  const unsigned gin = 1u << lgin;
  const int gmask = (1 << lgout) - 1;

  // per-lane output-voxel coords for the two M fragments
  int vx[2], vy[2], vz[2];
#pragma unroll
  for (int mb = 0; mb < 2; ++mb) {
    int lin = tile_base + mb * 16 + m;
    if (KSIZE > 1) {
      vz[mb] = lin & gmask;
      vy[mb] = (lin >> lgout) & gmask;
      vx[mb] = lin >> (2 * lgout);
    } else {
      vx[mb] = lin;  // linear row id for 1x1 / GEMM
    }
  }

  v8f acc[2][2];
#pragma unroll
  for (int mb = 0; mb < 2; ++mb)
#pragma unroll
    for (int cb = 0; cb < 2; ++cb)
#pragma unroll
      for (int i = 0; i < 8; ++i) acc[mb][cb][i] = 0.0f;

  for (int dx = 0; dx < KSIZE; ++dx)
    for (int dy = 0; dy < KSIZE; ++dy)
      for (int dz = 0; dz < KSIZE; ++dz) {
        const int tap = (dx * KSIZE + dy) * KSIZE + dz;
        bool valid[2];
        const __bf16* ap[2];
#pragma unroll
        for (int mb = 0; mb < 2; ++mb) {
          int in_vox;
          if (KSIZE == 1) {
            valid[mb] = true;
            in_vox = vx[mb];
          } else {
            int gx = vx[mb] * STRIDE + dx - PAD;
            int gy = vy[mb] * STRIDE + dy - PAD;
            int gz = vz[mb] * STRIDE + dz - PAD;
            valid[mb] = ((unsigned)gx < gin) & ((unsigned)gy < gin) &
                        ((unsigned)gz < gin);
            in_vox = valid[mb] ? ((((gx << lgin) | gy) << lgin) | gz) : 0;
          }
          ap[mb] = in + (size_t)in_vox * cin + khalf * 8;
        }
        for (int kc = 0; kc < nkc; ++kc) {
          // A fragments: 16-bit A 16x32 layout (row m, two 8-elem K runs)
          v16bf a[2];
#pragma unroll
          for (int mb = 0; mb < 2; ++mb) {
            v8bf lo, hi;
#pragma unroll
            for (int i = 0; i < 8; ++i) { lo[i] = (__bf16)0.0f; hi[i] = (__bf16)0.0f; }
            if (valid[mb]) {
              lo = *(const v8bf*)(ap[mb] + kc * 32);        // K = khalf*8 + [0..7]
              hi = *(const v8bf*)(ap[mb] + kc * 32 + 16);   // K = 16+khalf*8+[0..7]
            }
            a[mb] = __builtin_shufflevector(lo, hi, 0, 1, 2, 3, 4, 5, 6, 7,
                                            8, 9, 10, 11, 12, 13, 14, 15);
          }
          // B fragments: pre-packed, one contiguous 32B read per lane
          v16bf b[2];
#pragma unroll
          for (int cb = 0; cb < 2; ++cb) {
            const __bf16* bp = wpk +
                (((size_t)tap * nkc + kc) * nct + (ct0 + cb)) * 512 + lane * 16;
            b[cb] = *(const v16bf*)bp;
            __builtin_prefetch(bp + (size_t)nct * 512, 0, 0);
          }
#pragma unroll
          for (int mb = 0; mb < 2; ++mb)
#pragma unroll
            for (int cb = 0; cb < 2; ++cb)
              acc[mb][cb] = __builtin_amdgcn_wmma_f32_16x16x32_bf16(
                  false, a[mb], false, b[cb], (short)0, acc[mb][cb], false, false);
        }
      }

  // D layout: VGPR r -> row (r + 8*khalf), col = lane&15
#pragma unroll
  for (int mb = 0; mb < 2; ++mb)
#pragma unroll
    for (int cb = 0; cb < 2; ++cb)
#pragma unroll
      for (int r = 0; r < 8; ++r) {
        int orow = tile_base + mb * 16 + r + 8 * khalf;
        float mv = mask ? mask[orow] : 1.0f;
        out[(size_t)orow * coutpad + (ct0 + cb) * 16 + (lane & 15)] =
            acc[mb][cb][r] * mv;
      }
}

// ------------------------------- batch norm --------------------------------

__global__ void bn_stats_k(const float* __restrict__ y, float* __restrict__ sum,
                           float* __restrict__ sumsq, int v, int c) {
  int ci = threadIdx.x;
  float s = 0.0f, s2 = 0.0f;
  for (int vi = blockIdx.x; vi < v; vi += gridDim.x) {
    float x = y[(size_t)vi * c + ci];
    s += x;
    s2 += x * x;
  }
  atomicAdd(&sum[ci], s);
  atomicAdd(&sumsq[ci], s2);
}

// val = ((y - mean) * rsqrt(var+eps) * g + b) * mask [+ addend] [relu]
// addMode: 0 none, 1 bf16 activation (identity residual), 2 fp32 buffer
__global__ void bn_apply_k(const float* __restrict__ y, const float* __restrict__ mask,
                           const float* __restrict__ sum, const float* __restrict__ sumsq,
                           const float* __restrict__ nptr,
                           const float* __restrict__ g, const float* __restrict__ b,
                           int v, int c, int add_mode,
                           const __bf16* __restrict__ add_bf,
                           const float* __restrict__ add_f, int do_relu,
                           __bf16* __restrict__ out_bf, float* __restrict__ out_f) {
  size_t idx = (size_t)blockIdx.x * blockDim.x + threadIdx.x;
  if (idx >= (size_t)v * c) return;
  int ci = (int)(idx % c);
  size_t vi = idx / c;
  float n = fmaxf(nptr[0], 1.0f);
  float mean = sum[ci] / n;
  float var = sumsq[ci] / n - mean * mean;
  float val = (y[idx] - mean) * rsqrtf(var + BN_EPS) * g[ci] + b[ci];
  val *= mask[vi];
  if (add_mode == 1) val += (float)add_bf[idx];
  else if (add_mode == 2) val += add_f[idx];
  if (do_relu) val = fmaxf(val, 0.0f);
  if (out_bf) out_bf[idx] = (__bf16)val;
  else out_f[idx] = val;
}

// ------------------------------- trilinear ---------------------------------

__global__ void trilinear_k(const __bf16* __restrict__ grid, int g, int c,
                            const int* __restrict__ coords, float scale,
                            __bf16* __restrict__ z, int zoff) {
  int n = blockIdx.x;
  int ci = threadIdx.x;
  float px = coords[n * 3 + 0] * scale;
  float py = coords[n * 3 + 1] * scale;
  float pz = coords[n * 3 + 2] * scale;
  int ix = (int)floorf(px), iy = (int)floorf(py), iz = (int)floorf(pz);
  float fx = px - ix, fy = py - iy, fz = pz - iz;
  float acc = 0.0f;
#pragma unroll
  for (int dx = 0; dx < 2; ++dx)
#pragma unroll
    for (int dy = 0; dy < 2; ++dy)
#pragma unroll
      for (int dz = 0; dz < 2; ++dz) {
        int jx = ix + dx, jy = iy + dy, jz = iz + dz;
        bool ok = ((unsigned)jx < (unsigned)g) & ((unsigned)jy < (unsigned)g) &
                  ((unsigned)jz < (unsigned)g);
        float w = (dx ? fx : 1.0f - fx) * (dy ? fy : 1.0f - fy) *
                  (dz ? fz : 1.0f - fz);
        if (ok)
          acc += w * (float)grid[(size_t)((jx * g + jy) * g + jz) * c + ci];
      }
  z[(size_t)n * ZDIM + zoff + ci] = (__bf16)acc;
}

// ------------------------------- epilogue ----------------------------------

__global__ void bias_out_k(const float* __restrict__ y32,
                           const float* __restrict__ bias,
                           float* __restrict__ out, int n) {
  size_t idx = (size_t)blockIdx.x * blockDim.x + threadIdx.x;
  if (idx >= (size_t)n * NCLS) return;
  int pi = (int)(idx / NCLS), k = (int)(idx % NCLS);
  out[idx] = y32[(size_t)pi * NCLSPAD + k] + bias[k];
}

// ---------------------------------------------------------------------------
// Host orchestration
// ---------------------------------------------------------------------------

enum {
  IN_FEATS = 0, IN_STEM_W0, IN_STEM_G0, IN_STEM_B0, IN_STEM_W1, IN_STEM_G1, IN_STEM_B1,
  IN_S1_W, IN_S1_G, IN_S1_B, IN_S1_WA, IN_S1_GA, IN_S1_BA, IN_S1_WB, IN_S1_GB, IN_S1_BB,
  IN_S2_W, IN_S2_G, IN_S2_B, IN_S2_WA, IN_S2_GA, IN_S2_BA, IN_S2_WB, IN_S2_GB, IN_S2_BB,
  IN_S2_WD, IN_S2_GD, IN_S2_BD,
  IN_S3_W, IN_S3_G, IN_S3_B, IN_S3_WA, IN_S3_GA, IN_S3_BA, IN_S3_WB, IN_S3_GB, IN_S3_BB,
  IN_S4_W, IN_S4_G, IN_S4_B, IN_S4_WA, IN_S4_GA, IN_S4_BA, IN_S4_WB, IN_S4_GB, IN_S4_BB,
  IN_S4_WD, IN_S4_GD, IN_S4_BD,
  IN_CLS_W, IN_CLS_B, IN_COORDS
};

extern "C" void kernel_launch(void* const* d_in, const int* in_sizes, int n_in,
                              void* d_out, int out_size, void* d_ws, size_t ws_size,
                              hipStream_t stream) {
  (void)in_sizes; (void)n_in; (void)out_size; (void)ws_size;

  // ---- workspace bump allocator ----
  size_t off = 0;
  auto alloc = [&](size_t bytes) -> void* {
    void* p = (char*)d_ws + off;
    off += (bytes + 255) & ~(size_t)255;
    return p;
  };

  const int V64 = GDIM * GDIM * GDIM;          // 262144
  const int V32 = V64 / 8, V16 = V32 / 8, V8 = V16 / 8, V4 = V8 / 8;

  const size_t ACT_ELEMS = (size_t)V64 * 96;
  __bf16* actA = (__bf16*)alloc(ACT_ELEMS * 2);
  __bf16* actB = (__bf16*)alloc(ACT_ELEMS * 2);
  __bf16* actT = (__bf16*)alloc((size_t)V32 * 96 * 2);   // res-block temp
  float*  ybuf = (float*)alloc(ACT_ELEMS * 4);           // conv fp32 output
  float*  ydbuf = (float*)alloc((size_t)V16 * 128 * 4);  // shortcut conv output
  __bf16* zbuf = (__bf16*)alloc((size_t)NPTS * ZDIM * 2);
  float*  out32 = (float*)alloc((size_t)NPTS * NCLSPAD * 4);

  float* mask0 = (float*)alloc((size_t)V64 * 4);
  float* mask1 = (float*)alloc((size_t)V32 * 4);
  float* mask2 = (float*)alloc((size_t)V16 * 4);
  float* mask3 = (float*)alloc((size_t)V8 * 4);
  float* mask4 = (float*)alloc((size_t)V4 * 4);
  float* nsum = (float*)alloc(8 * 4);           // mask counts per level
  float* stat = (float*)alloc(512 * 4);         // sum[256] | sumsq[256]
  float* stat_sq = stat + 256;

  // ---- helpers ----
  auto zero_u32 = [&](void* p, size_t n32) {
    zero_u32_k<<<(unsigned)((n32 + 255) / 256), 256, 0, stream>>>((unsigned*)p, n32);
  };
  auto pack_w = [&](int idx, int ntap, int cin, int cout, int coutpad) -> __bf16* {
    size_t elems = (size_t)ntap * (cin / 32) * (coutpad / 16) * 512;
    __bf16* pk = (__bf16*)alloc(elems * 2);
    pack_w_k<<<(unsigned)((elems + 255) / 256), 256, 0, stream>>>(
        (const float*)d_in[idx], pk, ntap, cin, cout, coutpad);
    return pk;
  };
  auto lg2 = [](int g) { int l = 0; while ((1 << l) < g) ++l; return l; };
  auto conv = [&](const __bf16* in, const __bf16* pk, float* out, const float* mask,
                  int gin, int gout, int v, int cin, int coutpad, int k) {
    int pairs = coutpad / 32;                 // ctile-pairs (one per wave)
    dim3 grid(v / 32, (pairs + 1) / 2);       // 2 waves per 64-thread block
    if (k == 3)
      conv_wmma_k<3><<<grid, 64, 0, stream>>>(in, pk, out, mask, lg2(gin),
                                              lg2(gout), cin, coutpad);
    else if (k == 2)
      conv_wmma_k<2><<<grid, 64, 0, stream>>>(in, pk, out, mask, lg2(gin),
                                              lg2(gout), cin, coutpad);
    else
      conv_wmma_k<1><<<grid, 64, 0, stream>>>(in, pk, out, mask, 0, 0, cin,
                                              coutpad);
  };
  auto mask_count = [&](const float* m, int v, float* dst) {
    zero_u32(dst, 1);
    reduce_sum_k<<<256, 256, 0, stream>>>(m, v, dst);
  };
  auto bn = [&](const float* y, const float* mask, const float* nptr,
                int gidx, int bidx, int v, int c, int add_mode,
                const __bf16* add_bf, const float* add_f, int relu,
                __bf16* out_bf, float* out_f) {
    zero_u32(stat, 512);
    bn_stats_k<<<(v < 256 ? v : 256), c, 0, stream>>>(y, stat, stat_sq, v, c);
    size_t tot = (size_t)v * c;
    bn_apply_k<<<(unsigned)((tot + 255) / 256), 256, 0, stream>>>(
        y, mask, stat, stat_sq, nptr, (const float*)d_in[gidx],
        (const float*)d_in[bidx], v, c, add_mode, add_bf, add_f, relu,
        out_bf, out_f);
  };

  // ---- pack all weights to bf16 WMMA fragments ----
  __bf16* pk_w0 = pack_w(IN_STEM_W0, 27, 96, 96, 96);
  __bf16* pk_w1 = pack_w(IN_STEM_W1, 27, 96, 96, 96);
  __bf16* pk_s1w = pack_w(IN_S1_W, 8, 96, 96, 96);
  __bf16* pk_s1a = pack_w(IN_S1_WA, 27, 96, 96, 96);
  __bf16* pk_s1b = pack_w(IN_S1_WB, 27, 96, 96, 96);
  __bf16* pk_s2w = pack_w(IN_S2_W, 8, 96, 96, 96);
  __bf16* pk_s2a = pack_w(IN_S2_WA, 27, 96, 128, 128);
  __bf16* pk_s2b = pack_w(IN_S2_WB, 27, 128, 128, 128);
  __bf16* pk_s2d = pack_w(IN_S2_WD, 1, 96, 128, 128);
  __bf16* pk_s3w = pack_w(IN_S3_W, 8, 128, 128, 128);
  __bf16* pk_s3a = pack_w(IN_S3_WA, 27, 128, 128, 128);
  __bf16* pk_s3b = pack_w(IN_S3_WB, 27, 128, 128, 128);
  __bf16* pk_s4w = pack_w(IN_S4_W, 8, 128, 128, 128);
  __bf16* pk_s4a = pack_w(IN_S4_WA, 27, 128, 256, 256);
  __bf16* pk_s4b = pack_w(IN_S4_WB, 27, 256, 256, 256);
  __bf16* pk_s4d = pack_w(IN_S4_WD, 1, 128, 256, 256);
  __bf16* pk_cls = pack_w(IN_CLS_W, 1, ZDIM, NCLS, NCLSPAD);

  const int* coords = (const int*)d_in[IN_COORDS];

  // ---- scatter features, build masks ----
  zero_u32(actA, ACT_ELEMS / 2);
  zero_u32(mask0, V64);
  {
    size_t tot = (size_t)NPTS * CIN0;
    scatter_k<<<(unsigned)((tot + 255) / 256), 256, 0, stream>>>(
        (const float*)d_in[IN_FEATS], coords, actA, mask0, NPTS, CIN0);
  }
  mask_count(mask0, V64, &nsum[0]);

  // ---- stem ----
  conv(actA, pk_w0, ybuf, mask0, GDIM, GDIM, V64, 96, 96, 3);
  bn(ybuf, mask0, &nsum[0], IN_STEM_G0, IN_STEM_B0, V64, 96, 0, nullptr, nullptr, 1, actB, nullptr);
  conv(actB, pk_w1, ybuf, mask0, GDIM, GDIM, V64, 96, 96, 3);
  bn(ybuf, mask0, &nsum[0], IN_STEM_G1, IN_STEM_B1, V64, 96, 0, nullptr, nullptr, 1, actA, nullptr);  // x0
  trilinear_k<<<NPTS, 96, 0, stream>>>(actA, GDIM, 96, coords, 1.0f, zbuf, 0);   // z0

  // ---- stage 1 (64 -> 32, 96ch) ----
  pool_mask_k<<<(V32 + 255) / 256, 256, 0, stream>>>(mask0, mask1, 5);
  mask_count(mask1, V32, &nsum[1]);
  conv(actA, pk_s1w, ybuf, mask1, GDIM, 32, V32, 96, 96, 2);
  bn(ybuf, mask1, &nsum[1], IN_S1_G, IN_S1_B, V32, 96, 0, nullptr, nullptr, 1, actB, nullptr);
  conv(actB, pk_s1a, ybuf, mask1, 32, 32, V32, 96, 96, 3);
  bn(ybuf, mask1, &nsum[1], IN_S1_GA, IN_S1_BA, V32, 96, 0, nullptr, nullptr, 1, actT, nullptr);
  conv(actT, pk_s1b, ybuf, mask1, 32, 32, V32, 96, 96, 3);
  bn(ybuf, mask1, &nsum[1], IN_S1_GB, IN_S1_BB, V32, 96, 1, actB, nullptr, 1, actA, nullptr);  // x1

  // ---- stage 2 (32 -> 16, 96 -> 128ch, downsample shortcut) ----
  pool_mask_k<<<(V16 + 255) / 256, 256, 0, stream>>>(mask1, mask2, 4);
  mask_count(mask2, V16, &nsum[2]);
  conv(actA, pk_s2w, ybuf, mask2, 32, 16, V16, 96, 96, 2);
  bn(ybuf, mask2, &nsum[2], IN_S2_G, IN_S2_B, V16, 96, 0, nullptr, nullptr, 1, actB, nullptr);
  conv(actB, pk_s2a, ybuf, mask2, 16, 16, V16, 96, 128, 3);
  bn(ybuf, mask2, &nsum[2], IN_S2_GA, IN_S2_BA, V16, 128, 0, nullptr, nullptr, 1, actT, nullptr);
  conv(actT, pk_s2b, ybuf, mask2, 16, 16, V16, 128, 128, 3);
  conv(actB, pk_s2d, ydbuf, mask2, 16, 16, V16, 96, 128, 1);
  bn(ydbuf, mask2, &nsum[2], IN_S2_GD, IN_S2_BD, V16, 128, 0, nullptr, nullptr, 0, nullptr, ydbuf);
  bn(ybuf, mask2, &nsum[2], IN_S2_GB, IN_S2_BB, V16, 128, 2, nullptr, ydbuf, 1, actA, nullptr);  // x2
  trilinear_k<<<NPTS, 128, 0, stream>>>(actA, 16, 128, coords, 0.25f, zbuf, 96);  // z1

  // ---- stage 3 (16 -> 8, 128ch) ----
  pool_mask_k<<<(V8 + 255) / 256, 256, 0, stream>>>(mask2, mask3, 3);
  mask_count(mask3, V8, &nsum[3]);
  conv(actA, pk_s3w, ybuf, mask3, 16, 8, V8, 128, 128, 2);
  bn(ybuf, mask3, &nsum[3], IN_S3_G, IN_S3_B, V8, 128, 0, nullptr, nullptr, 1, actB, nullptr);
  conv(actB, pk_s3a, ybuf, mask3, 8, 8, V8, 128, 128, 3);
  bn(ybuf, mask3, &nsum[3], IN_S3_GA, IN_S3_BA, V8, 128, 0, nullptr, nullptr, 1, actT, nullptr);
  conv(actT, pk_s3b, ybuf, mask3, 8, 8, V8, 128, 128, 3);
  bn(ybuf, mask3, &nsum[3], IN_S3_GB, IN_S3_BB, V8, 128, 1, actB, nullptr, 1, actA, nullptr);  // x3

  // ---- stage 4 (8 -> 4, 128 -> 256ch, downsample shortcut) ----
  pool_mask_k<<<1, 64, 0, stream>>>(mask3, mask4, 2);
  mask_count(mask4, V4, &nsum[4]);
  conv(actA, pk_s4w, ybuf, mask4, 8, 4, V4, 128, 128, 2);
  bn(ybuf, mask4, &nsum[4], IN_S4_G, IN_S4_B, V4, 128, 0, nullptr, nullptr, 1, actB, nullptr);
  conv(actB, pk_s4a, ybuf, mask4, 4, 4, V4, 128, 256, 3);
  bn(ybuf, mask4, &nsum[4], IN_S4_GA, IN_S4_BA, V4, 256, 0, nullptr, nullptr, 1, actT, nullptr);
  conv(actT, pk_s4b, ybuf, mask4, 4, 4, V4, 256, 256, 3);
  conv(actB, pk_s4d, ydbuf, mask4, 4, 4, V4, 128, 256, 1);
  bn(ydbuf, mask4, &nsum[4], IN_S4_GD, IN_S4_BD, V4, 256, 0, nullptr, nullptr, 0, nullptr, ydbuf);
  bn(ybuf, mask4, &nsum[4], IN_S4_GB, IN_S4_BB, V4, 256, 2, nullptr, ydbuf, 1, actA, nullptr);  // x4
  trilinear_k<<<NPTS, 256, 0, stream>>>(actA, 4, 256, coords, 0.0625f, zbuf, 224);  // z2

  // ---- classifier: [N x 480] @ [480 x 20] + bias (WMMA GEMM, Cout pad 32) ----
  conv(zbuf, pk_cls, out32, nullptr, 1, 1, NPTS, ZDIM, NCLSPAD, 1);
  {
    size_t tot = (size_t)NPTS * NCLS;
    bias_out_k<<<(unsigned)((tot + 255) / 256), 256, 0, stream>>>(
        out32, (const float*)d_in[IN_CLS_B], (float*)d_out, NPTS);
  }
}
